// GlimpseNet_8237747273859
// MI455X (gfx1250) — matrix operand order repaired and checked by reading, not compiled, not run on hardware
//
#include <hip/hip_runtime.h>

// ---------------------------------------------------------------------------
// GlimpseNet forward for MI455X (gfx1250, wave32, WMMA bf16 16x16x32).
//
// Roofline: f32 weights dominate traffic (W_img 192MB + W_fc1 80MB + W_fc2
// 20MB) -> ~12.5us at 23.3 TB/s. All GEMMs are weight-streaming bound
// (M=256), so weights are streamed once with non-temporal loads, converted
// to bf16 in registers, and fed to v_wmma_f32_16x16x32_bf16 (f32 accum).
// Activations are tiny bf16 matrices kept hot in the 192MB L2.
//
// flat = [patch|patch|patch] is never materialized; instead GEMM1 is
// SPLIT-K over the 3 replicated blocks (grid.z = 3) and GEMM2 is split 3
// ways (2 halves of the fc1 K-loop + the fc2 branch), with partials
// combined via relaxed atomic f32 adds. This triples wave count
// (3072 / 3840 waves) so HBM latency can actually be hidden at 23.3 TB/s.
// ---------------------------------------------------------------------------

typedef __attribute__((ext_vector_type(16))) __bf16       bf16x16;
typedef __attribute__((ext_vector_type(8)))  float        f32x8;
typedef __attribute__((ext_vector_type(4)))  unsigned int u32x4;

union AFrag { bf16x16 v; u32x4 q[2]; };

// A-matrix fragment (16x32 bf16, row-major memory).
// Layout: lanes 0-15: V0-7 = K[0..15] of row M=lane;
//         lanes 16-31: V0-7 = K[16..31].
__device__ __forceinline__ void load_a_frag(const __bf16* __restrict__ p, AFrag& f) {
  f.q[0] = *(const u32x4*)(p);
  f.q[1] = *(const u32x4*)(p + 8);
}

// B-matrix fragment (32x16, f32 weights row-major, leading dim ldn).
// Layout used here: lanes 0-15 hold K=0..15 of column N=lane (paired per VGPR),
// lanes 16-31 hold K=16..31. Strided dword loads coalesce across each
// 16-lane half (64B per k-row). Non-temporal: each weight is used once.
__device__ __forceinline__ bf16x16 load_b_frag_f32(const float* __restrict__ wp,
                                                   int ldn) {
  bf16x16 b;
#pragma unroll
  for (int j = 0; j < 8; ++j) {
    float x0 = __builtin_nontemporal_load(wp + (size_t)(2 * j)     * ldn);
    float x1 = __builtin_nontemporal_load(wp + (size_t)(2 * j + 1) * ldn);
    b[2 * j]     = (__bf16)x0;
    b[2 * j + 1] = (__bf16)x1;
  }
  return b;
}

__device__ __forceinline__ f32x8 zero8() {
  f32x8 z = {0.f, 0.f, 0.f, 0.f, 0.f, 0.f, 0.f, 0.f};
  return z;
}

__device__ __forceinline__ void atomic_add_f32(float* p, float v) {
  (void)__hip_atomic_fetch_add(p, v, __ATOMIC_RELAXED, __HIP_MEMORY_SCOPE_AGENT);
}

// Shared 32x32-per-wave WMMA tile state.
struct WaveTile {
  int lane, wave, m0, n0, half, l15;
  int aRow, aK, bN, bK;
  __device__ __forceinline__ void init(int mBlk, int nBlk) {
    lane = threadIdx.x & 31;
    wave = threadIdx.x >> 5;
    m0 = mBlk * 32;
    n0 = nBlk * 256 + wave * 32;
    half = lane >> 4;
    l15 = lane & 15;
    aRow = m0 + l15;
    aK = half * 16;
    bN = l15;
    bK = half * 16;
  }
};

// ---------------------------------------------------------------------------
__global__ __launch_bounds__(256) void zero_kernel(float* __restrict__ p, int n) {
  const int i = blockIdx.x * blockDim.x + threadIdx.x;
  if (i < n) p[i] = 0.0f;
}

// ---------------------------------------------------------------------------
// Patch extraction: loc_d = (loc+1)*128; masked gather from channel 0.
// Stored bf16 (256 x 4096).
// ---------------------------------------------------------------------------
__global__ __launch_bounds__(256) void patch_kernel(
    const float* __restrict__ img, const float* __restrict__ loc,
    __bf16* __restrict__ P) {
  const int b = blockIdx.x;
  const float xd = (loc[b * 2 + 0] + 1.0f) * 128.0f;
  const float yd = (loc[b * 2 + 1] + 1.0f) * 128.0f;
  const float* img0 = img + (size_t)b * 3 * 256 * 256;  // channel 0
  for (int t = threadIdx.x; t < 4096; t += blockDim.x) {
    const int i = t >> 6, j = t & 63;
    const float xf = xd + (float)i;
    const float yf = yd + (float)j;
    float v = 0.0f;
    if (xf < 255.0f && yf < 255.0f) {
      int ix = (int)xf; ix = ix < 0 ? 0 : (ix > 255 ? 255 : ix);
      int iy = (int)yf; iy = iy < 0 ? 0 : (iy > 255 ? 255 : iy);
      v = img0[ix * 256 + iy];
    }
    P[(size_t)b * 4096 + t] = (__bf16)v;
  }
}

// ---------------------------------------------------------------------------
// Location MLP: out_l = relu(loc_d @ W_loc + b_loc), K=2 -> plain VALU.
// Stored bf16 (256 x 1024).
// ---------------------------------------------------------------------------
__global__ __launch_bounds__(256) void loc_kernel(
    const float* __restrict__ loc, const float* __restrict__ Wl,
    const float* __restrict__ bl, __bf16* __restrict__ L) {
  const int idx = blockIdx.x * blockDim.x + threadIdx.x;  // 256*1024
  const int b = idx >> 10, j = idx & 1023;
  const float xd = (loc[b * 2 + 0] + 1.0f) * 128.0f;
  const float yd = (loc[b * 2 + 1] + 1.0f) * 128.0f;
  const float v = fmaf(xd, Wl[j], fmaf(yd, Wl[1024 + j], bl[j]));
  L[idx] = (__bf16)fmaxf(v, 0.0f);
}

// ---------------------------------------------------------------------------
// GEMM1 split-K: partial of flat @ W_img into f32 accumulator via atomics.
// A = patch bf16 (256x4096); grid (8, 16, 3), z = replicated block index.
// Each wave: 2x2 WMMA tiles, 128 k-steps per split.
// ---------------------------------------------------------------------------
__global__ __launch_bounds__(256) void gemm1_kernel(
    const __bf16* __restrict__ A, const float* __restrict__ W,
    float* __restrict__ Acc) {
  WaveTile t; t.init(blockIdx.x, blockIdx.y);
  const int kt0 = blockIdx.z * 128;

  f32x8 acc[2][2];
#pragma unroll
  for (int mi = 0; mi < 2; ++mi)
#pragma unroll
    for (int ni = 0; ni < 2; ++ni) acc[mi][ni] = zero8();

#pragma unroll 2
  for (int kt = kt0; kt < kt0 + 128; ++kt) {
    const int k = kt * 32;
    const int ka = k & 4095;  // replicated-patch indexing
    AFrag a[2];
#pragma unroll
    for (int mi = 0; mi < 2; ++mi)
      load_a_frag(A + (size_t)(t.aRow + 16 * mi) * 4096 + ka + t.aK, a[mi]);
    bf16x16 b[2];
#pragma unroll
    for (int ni = 0; ni < 2; ++ni)
      b[ni] = load_b_frag_f32(W + (size_t)(k + t.bK) * 4096 + t.n0 + 16 * ni + t.bN, 4096);
#pragma unroll
    for (int mi = 0; mi < 2; ++mi)
#pragma unroll
      for (int ni = 0; ni < 2; ++ni)
        acc[mi][ni] = __builtin_amdgcn_wmma_f32_16x16x32_bf16(
            false, a[mi].v, false, b[ni], (short)0, acc[mi][ni], false, false);
  }

  // C/D layout: VGPR r -> (M = r + 8*half, N = l15) within the 16x16 tile.
#pragma unroll
  for (int mi = 0; mi < 2; ++mi)
#pragma unroll
    for (int ni = 0; ni < 2; ++ni)
#pragma unroll
      for (int r = 0; r < 8; ++r) {
        const int row = t.m0 + mi * 16 + r + t.half * 8;
        const int col = t.n0 + ni * 16 + t.l15;
        atomic_add_f32(Acc + (size_t)row * 4096 + col, acc[mi][ni][r]);
      }
}

// ---------------------------------------------------------------------------
// Epilogue 1: out_i = relu(acc + b_img) -> bf16 for GEMM2.
// ---------------------------------------------------------------------------
__global__ __launch_bounds__(256) void epi1_kernel(
    const float* __restrict__ Acc, const float* __restrict__ bias,
    __bf16* __restrict__ OI) {
  const int idx = blockIdx.x * blockDim.x + threadIdx.x;  // 256*4096
  const int col = idx & 4095;
  OI[idx] = (__bf16)fmaxf(Acc[idx] + bias[col], 0.0f);
}

// ---------------------------------------------------------------------------
// GEMM2 split: grid (8, 20, 3).
//   z=0: out_i[:,    0:2048] @ W_fc1[    0:2048]
//   z=1: out_i[:, 2048:4096] @ W_fc1[2048:4096]
//   z=2: out_l @ W_fc2
// Partials accumulated into d_out (f32) via atomics.
// ---------------------------------------------------------------------------
__global__ __launch_bounds__(256) void gemm2_kernel(
    const __bf16* __restrict__ A1, const __bf16* __restrict__ A2,
    const float* __restrict__ W1, const float* __restrict__ W2,
    float* __restrict__ Out) {
  WaveTile t; t.init(blockIdx.x, blockIdx.y);
  const int z = blockIdx.z;

  f32x8 acc[2][2];
#pragma unroll
  for (int mi = 0; mi < 2; ++mi)
#pragma unroll
    for (int ni = 0; ni < 2; ++ni) acc[mi][ni] = zero8();

  if (z < 2) {
    const int kt0 = z * 64;
#pragma unroll 2
    for (int kt = kt0; kt < kt0 + 64; ++kt) {
      const int k = kt * 32;
      AFrag a[2];
#pragma unroll
      for (int mi = 0; mi < 2; ++mi)
        load_a_frag(A1 + (size_t)(t.aRow + 16 * mi) * 4096 + k + t.aK, a[mi]);
      bf16x16 b[2];
#pragma unroll
      for (int ni = 0; ni < 2; ++ni)
        b[ni] = load_b_frag_f32(W1 + (size_t)(k + t.bK) * 5120 + t.n0 + 16 * ni + t.bN, 5120);
#pragma unroll
      for (int mi = 0; mi < 2; ++mi)
#pragma unroll
        for (int ni = 0; ni < 2; ++ni)
          acc[mi][ni] = __builtin_amdgcn_wmma_f32_16x16x32_bf16(
              false, a[mi].v, false, b[ni], (short)0, acc[mi][ni], false, false);
    }
  } else {
#pragma unroll 2
    for (int kt = 0; kt < 32; ++kt) {
      const int k = kt * 32;
      AFrag a[2];
#pragma unroll
      for (int mi = 0; mi < 2; ++mi)
        load_a_frag(A2 + (size_t)(t.aRow + 16 * mi) * 1024 + k + t.aK, a[mi]);
      bf16x16 b[2];
#pragma unroll
      for (int ni = 0; ni < 2; ++ni)
        b[ni] = load_b_frag_f32(W2 + (size_t)(k + t.bK) * 5120 + t.n0 + 16 * ni + t.bN, 5120);
#pragma unroll
      for (int mi = 0; mi < 2; ++mi)
#pragma unroll
        for (int ni = 0; ni < 2; ++ni)
          acc[mi][ni] = __builtin_amdgcn_wmma_f32_16x16x32_bf16(
              false, a[mi].v, false, b[ni], (short)0, acc[mi][ni], false, false);
    }
  }

#pragma unroll
  for (int mi = 0; mi < 2; ++mi)
#pragma unroll
    for (int ni = 0; ni < 2; ++ni)
#pragma unroll
      for (int r = 0; r < 8; ++r) {
        const int row = t.m0 + mi * 16 + r + t.half * 8;
        const int col = t.n0 + ni * 16 + t.l15;
        atomic_add_f32(Out + (size_t)row * 5120 + col, acc[mi][ni][r]);
      }
}

// ---------------------------------------------------------------------------
// Epilogue 2: out = relu(out + b_fc1 + b_fc2) in place on d_out.
// ---------------------------------------------------------------------------
__global__ __launch_bounds__(256) void epi2_kernel(
    float* __restrict__ Out, const float* __restrict__ b1,
    const float* __restrict__ b2) {
  const int idx = blockIdx.x * blockDim.x + threadIdx.x;  // 256*5120
  const int col = idx % 5120;
  Out[idx] = fmaxf(Out[idx] + b1[col] + b2[col], 0.0f);
}

// ---------------------------------------------------------------------------
extern "C" void kernel_launch(void* const* d_in, const int* in_sizes, int n_in,
                              void* d_out, int out_size, void* d_ws, size_t ws_size,
                              hipStream_t stream) {
  (void)in_sizes; (void)n_in; (void)out_size; (void)ws_size;
  const float* img  = (const float*)d_in[0];
  const float* loc  = (const float*)d_in[1];
  const float* Wimg = (const float*)d_in[2];
  const float* bimg = (const float*)d_in[3];
  const float* Wloc = (const float*)d_in[4];
  const float* bloc = (const float*)d_in[5];
  const float* Wfc1 = (const float*)d_in[6];
  const float* bfc1 = (const float*)d_in[7];
  const float* Wfc2 = (const float*)d_in[8];
  const float* bfc2 = (const float*)d_in[9];
  float* out = (float*)d_out;

  // Workspace: P bf16 2MB | L bf16 0.5MB | OI bf16 2MB | OIacc f32 4MB
  __bf16* P     = (__bf16*)d_ws;
  __bf16* L     = P + (size_t)256 * 4096;
  __bf16* OI    = L + (size_t)256 * 1024;
  float*  OIacc = (float*)(OI + (size_t)256 * 4096);

  const int nOI  = 256 * 4096;   // 1,048,576
  const int nOut = 256 * 5120;   // 1,310,720

  zero_kernel<<<(nOI  + 255) / 256, 256, 0, stream>>>(OIacc, nOI);
  zero_kernel<<<(nOut + 255) / 256, 256, 0, stream>>>(out, nOut);
  patch_kernel<<<256, 256, 0, stream>>>(img, loc, P);
  loc_kernel<<<1024, 256, 0, stream>>>(loc, Wloc, bloc, L);

  gemm1_kernel<<<dim3(8, 16, 3), 256, 0, stream>>>(P, Wimg, OIacc);
  epi1_kernel<<<nOI / 256, 256, 0, stream>>>(OIacc, bimg, OI);

  gemm2_kernel<<<dim3(8, 20, 3), 256, 0, stream>>>(OI, L, Wfc1, Wfc2, out);
  epi2_kernel<<<nOut / 256, 256, 0, stream>>>(out, bfc1, bfc2);
}
